// VSAWindowAttention_33646773796920
// MI455X (gfx1250) — matrix operand order repaired and checked by reading, not compiled, not run on hardware
//
#include <hip/hip_runtime.h>

typedef __attribute__((ext_vector_type(16))) _Float16 v16h;
typedef __attribute__((ext_vector_type(8)))  _Float16 v8h;
typedef __attribute__((ext_vector_type(8)))  float    v8f;

#define B_     16
#define DIM_   96
#define HW_    112
#define NPIX   (HW_*HW_)
#define HEADS_ 3
#define WS_    7
#define WIN_   49
#define DH_    32
#define WN_    16
#define NWIN_  (B_*WN_*WN_)
#define NTILES (B_*NPIX/16)   /* 12544 N-tiles of 16 pixels */

static __device__ __forceinline__ v8f wmma_f16(v16h a, v16h b, v8f c) {
  return __builtin_amdgcn_wmma_f32_16x16x32_f16(false, a, false, b, (short)0, c, false, false);
}

// ---------------------------------------------------------------------------
// Kernel 1: qkv = w_qkv (288x96) @ x (96 x 200704) + b_qkv -> f16
// Block: 8 waves, all working on one "which" group (q/k/v, 96 rows).
// Weights for the group staged to LDS as f16 once per block.
// Each wave owns one N-tile (16 pixels) and iterates the 6 M-tiles,
// holding the B fragment in registers across M (x read only 3x total).
// ---------------------------------------------------------------------------
__global__ void __launch_bounds__(256) qkv_kernel(
    const float* __restrict__ x, const float* __restrict__ w_qkv,
    const float* __restrict__ b_qkv,
    _Float16* __restrict__ qbuf, _Float16* __restrict__ kbuf,
    _Float16* __restrict__ vbuf)
{
  __shared__ _Float16 wlds[96][96];          // 18 KB, A rows for this group

  const int which = blockIdx.x / (NTILES / 8);     // 0=q 1=k 2=v
  const int nblk  = blockIdx.x % (NTILES / 8);
  const int tid   = threadIdx.x;

  // stage 96x96 f32 weight block -> f16 LDS (row-major [m][k])
  {
    const float* wsrc = w_qkv + (size_t)which * 96 * DIM_;
    _Float16* wd = &wlds[0][0];
    for (int i = tid; i < 96 * 96; i += 256) wd[i] = (_Float16)wsrc[i];
  }
  __syncthreads();

  const int lane = tid & 31;
  const int wv   = tid >> 5;
  const int ln = lane & 15, lh = lane >> 4;
  const int nt = nblk * 8 + wv;
  const int p  = nt * 16 + ln;
  const int b  = p / NPIX, yx = p % NPIX;
  const float* xb = x + (size_t)b * DIM_ * NPIX + yx;

  _Float16* dstbase = (which == 0) ? qbuf : (which == 1) ? kbuf : vbuf;

  v8f acc[6] = {{}, {}, {}, {}, {}, {}};
  #pragma unroll
  for (int ks = 0; ks < 3; ++ks) {
    v16h bm;
    #pragma unroll
    for (int j = 0; j < 16; ++j) {       // B layout: K = j + 16*lh
      const int c = ks * 32 + (lh << 4) + j;
      bm[j] = (_Float16)xb[(size_t)c * NPIX];
    }
    #pragma unroll
    for (int mtl = 0; mtl < 6; ++mtl) {  // A layout: K = j + 16*lh
      const _Float16* ar = &wlds[mtl * 16 + ln][ks * 32];
      v8h alo = *(const v8h*)(ar + 16 * lh);
      v8h ahi = *(const v8h*)(ar + 16 * lh + 8);
      v16h a;
      #pragma unroll
      for (int j = 0; j < 8; ++j) { a[j] = alo[j]; a[j + 8] = ahi[j]; }
      acc[mtl] = wmma_f16(a, bm, acc[mtl]);
    }
  }

  #pragma unroll
  for (int mtl = 0; mtl < 6; ++mtl) {
    const int mloc  = mtl * 16 + lh * 8;          // row inside the 96 group
    const int head  = mloc / 32, dch = mloc % 32; // 8-aligned: one head slab
    const int mglob = which * 96 + mloc;
    _Float16* dst = dstbase + ((size_t)(b * HEADS_ + head) * NPIX + yx) * DH_ + dch;
    v8h o;
    #pragma unroll
    for (int r = 0; r < 8; ++r) o[r] = (_Float16)(acc[mtl][r] + b_qkv[mglob + r]);
    *(v8h*)dst = o;
  }
}

// ---------------------------------------------------------------------------
// Kernel 2: window mean-pool -> leaky_relu -> off / scl projections (tiny)
// ---------------------------------------------------------------------------
__global__ void pool_off_scl_kernel(
    const float* __restrict__ x,
    const float* __restrict__ w_off, const float* __restrict__ b_off,
    const float* __restrict__ w_scl, const float* __restrict__ b_scl,
    float* __restrict__ offBuf, float* __restrict__ sclBuf)
{
  __shared__ float acts[DIM_];
  const int cell = blockIdx.x;
  const int b  = cell >> 8;
  const int wy = (cell >> 4) & 15;
  const int wx = cell & 15;
  const int c = threadIdx.x;
  const float* xb = x + ((size_t)b * DIM_ + c) * NPIX + (wy * WS_) * HW_ + wx * WS_;
  float s = 0.f;
  #pragma unroll
  for (int ty = 0; ty < WS_; ++ty)
    #pragma unroll
    for (int tx = 0; tx < WS_; ++tx)
      s += xb[ty * HW_ + tx];
  s *= (1.f / 49.f);
  acts[c] = (s > 0.f) ? s : 0.01f * s;   // leaky_relu(0.01)
  __syncthreads();
  if (c < 12) {
    const int isScl = (c >= 6);
    const int o = c % 6;                 // o = head*2 + comp
    const float* w = (isScl ? w_scl : w_off) + o * DIM_;
    float d = (isScl ? b_scl : b_off)[o];
    for (int i = 0; i < DIM_; ++i) d += acts[i] * w[i];
    if (!isScl) d *= (1.f / 16.f);       // off /= (h//ws) == 16 (both comps)
    float* dst = isScl ? sclBuf : offBuf;
    dst[(size_t)(b * 6 + o) * 256 + wy * 16 + wx] = d;
  }
}

// ---------------------------------------------------------------------------
// Kernel 3: fused bilinear sampling + windowed attention.
// Block = one (window, head): 4 waves, wave w owns q-tile w (16 queries).
// rpb bias table staged to LDS via CDNA5 async global->LDS loads.
// ---------------------------------------------------------------------------
__global__ void __launch_bounds__(128) attn_kernel(
    const _Float16* __restrict__ qbuf, const _Float16* __restrict__ kbuf,
    const _Float16* __restrict__ vbuf,
    const float* __restrict__ offBuf, const float* __restrict__ sclBuf,
    const float* __restrict__ rpb, _Float16* __restrict__ aout)
{
  __shared__ _Float16 kl[64][DH_];      // sampled K: [key][dh]
  __shared__ _Float16 vlt[DH_][64];     // sampled V transposed: [dh][key]
  __shared__ _Float16 pl[4][16][64];    // softmax(P) staging per q-tile
  __shared__ float rpb_lds[512];        // 169*3 = 507 bias values

  const int unit = blockIdx.x;
  const int hd = unit % HEADS_;
  const int cell = unit / HEADS_;
  const int b  = cell >> 8;
  const int wy = (cell >> 4) & 15;
  const int wx = cell & 15;
  const int tid = threadIdx.x;
  const int bh = b * HEADS_ + hd;

  // --- async copy rpb table into LDS (ASYNCcnt-tracked, gfx1250 path) ---
  #pragma unroll
  for (int i = 0; i < 4; ++i) {
    const int idx = tid + i * 128;
    if (idx < 169 * HEADS_) {
      const unsigned lofs = (unsigned)(size_t)(&rpb_lds[idx]);
      const unsigned long long ga = (unsigned long long)(size_t)(rpb + idx);
      asm volatile("global_load_async_to_lds_b32 %0, %1, off"
                   :: "v"(lofs), "v"(ga) : "memory");
    }
  }

  {
    _Float16* kz = &kl[0][0];
    _Float16* vz = &vlt[0][0];
    for (int i = tid; i < 64 * DH_; i += 128) {
      kz[i] = (_Float16)0.f; vz[i] = (_Float16)0.f;
    }
  }
  asm volatile("s_wait_asynccnt 0" ::: "memory");
  __syncthreads();

  // --- bilinear sampling of K,V at deformed grid (one token per thread) ---
  if (tid < WIN_) {
    const int ty = tid / WS_, tx = tid % WS_;
    const float sx = sclBuf[((size_t)bh * 2 + 0) * 256 + wy * 16 + wx];
    const float sy = sclBuf[((size_t)bh * 2 + 1) * 256 + wy * 16 + wx];
    const float ox = offBuf[((size_t)bh * 2 + 0) * 256 + wy * 16 + wx];
    const float oy = offBuf[((size_t)bh * 2 + 1) * 256 + wy * 16 + wx];
    const float gx = -1.f + 2.f * (float)(wx * WS_ + tx) * (1.f / 111.f)
                   + (float)(tx - 3) * (2.f / 111.f) * sx + ox;
    const float gy = -1.f + 2.f * (float)(wy * WS_ + ty) * (1.f / 111.f)
                   + (float)(ty - 3) * (2.f / 111.f) * sy + oy;
    const float px = (gx + 1.f) * 0.5f * 111.f;
    const float py = (gy + 1.f) * 0.5f * 111.f;
    const float x0f = floorf(px), y0f = floorf(py);
    const int x0 = (int)x0f, y0 = (int)y0f;
    const float fx = px - x0f, fy = py - y0f;
    const bool va = (x0   >= 0 && x0   < HW_ && y0   >= 0 && y0   < HW_);
    const bool vb = (x0+1 >= 0 && x0+1 < HW_ && y0   >= 0 && y0   < HW_);
    const bool vc = (x0   >= 0 && x0   < HW_ && y0+1 >= 0 && y0+1 < HW_);
    const bool vd = (x0+1 >= 0 && x0+1 < HW_ && y0+1 >= 0 && y0+1 < HW_);
    const float wa2 = va ? (1.f - fx) * (1.f - fy) : 0.f;
    const float wb2 = vb ? fx * (1.f - fy) : 0.f;
    const float wc2 = vc ? (1.f - fx) * fy : 0.f;
    const float wd2 = vd ? fx * fy : 0.f;
    const int xa = min(max(x0, 0), HW_ - 1),  ya = min(max(y0, 0), HW_ - 1);
    const int xb2 = min(max(x0 + 1, 0), HW_ - 1), yb2 = min(max(y0 + 1, 0), HW_ - 1);
    const _Float16* kb = kbuf + (size_t)bh * NPIX * DH_;
    const _Float16* vp = vbuf + (size_t)bh * NPIX * DH_;
    const size_t i00 = ((size_t)ya  * HW_ + xa ) * DH_;
    const size_t i10 = ((size_t)ya  * HW_ + xb2) * DH_;
    const size_t i01 = ((size_t)yb2 * HW_ + xa ) * DH_;
    const size_t i11 = ((size_t)yb2 * HW_ + xb2) * DH_;
    #pragma unroll
    for (int d = 0; d < DH_; ++d) {
      const float kv = wa2 * (float)kb[i00+d] + wb2 * (float)kb[i10+d]
                     + wc2 * (float)kb[i01+d] + wd2 * (float)kb[i11+d];
      const float vv = wa2 * (float)vp[i00+d] + wb2 * (float)vp[i10+d]
                     + wc2 * (float)vp[i01+d] + wd2 * (float)vp[i11+d];
      kl[tid][d]  = (_Float16)kv;
      vlt[d][tid] = (_Float16)vv;
    }
  }
  __syncthreads();

  const int lane = tid & 31;
  const int wvq  = tid >> 5;            // q-tile 0..3
  const int ln = lane & 15, lh = lane >> 4;

  // A = Q rows (token per lane, 32 contiguous halves per token)
  v16h aq = {};
  const int tq = wvq * 16 + ln;
  if (tq < WIN_) {
    const int qy = tq / WS_, qx = tq % WS_;
    const _Float16* qrow = qbuf
        + ((size_t)bh * NPIX + (size_t)(wy * WS_ + qy) * HW_ + (wx * WS_ + qx)) * DH_;
    v8h lo = *(const v8h*)(qrow + 16 * lh);
    v8h hi = *(const v8h*)(qrow + 16 * lh + 8);
    #pragma unroll
    for (int j = 0; j < 8; ++j) { aq[j] = lo[j]; aq[j + 8] = hi[j]; }
  }

  // S = Q K^T : 4 WMMAs (dh=32 -> exactly one K step each)
  v8f s[4];
  #pragma unroll
  for (int kt = 0; kt < 4; ++kt) {
    v16h bm = *(const v16h*)&kl[kt * 16 + ln][16 * lh];
    v8f z = {};
    s[kt] = wmma_f16(aq, bm, z);
  }

  // softmax over 49 keys (rows split across lane-halves; shfl stays in-half)
  const float scale = 0.17677669529663687f;   // 32^-0.5
  #pragma unroll
  for (int r = 0; r < 8; ++r) {
    const int qq = wvq * 16 + r + 8 * lh;
    const int qy = qq / WS_, qx = qq % WS_;
    float sv[4]; float mx = -1e30f;
    #pragma unroll
    for (int kt = 0; kt < 4; ++kt) {
      const int k2 = kt * 16 + ln;
      float val = -1e30f;
      if (qq < WIN_ && k2 < WIN_) {
        const int ky = k2 / WS_, kx = k2 % WS_;
        const int ridx = (qy - ky + 6) * 13 + (qx - kx + 6);
        val = s[kt][r] * scale + rpb_lds[ridx * HEADS_ + hd];
      }
      sv[kt] = val; mx = fmaxf(mx, val);
    }
    #pragma unroll
    for (int m = 1; m <= 8; m <<= 1) mx = fmaxf(mx, __shfl_xor(mx, m, 32));
    float sum = 0.f;
    #pragma unroll
    for (int kt = 0; kt < 4; ++kt) {
      const float e = (sv[kt] <= -1e29f) ? 0.f : __expf(sv[kt] - mx);
      sv[kt] = e; sum += e;
    }
    #pragma unroll
    for (int m = 1; m <= 8; m <<= 1) sum += __shfl_xor(sum, m, 32);
    const float inv = (sum > 0.f) ? 1.f / sum : 0.f;
    #pragma unroll
    for (int kt = 0; kt < 4; ++kt)
      pl[wvq][r + 8 * lh][kt * 16 + ln] = (_Float16)(sv[kt] * inv);
  }
  __syncthreads();

  // O = P V : K=64 in 2 steps x 2 n-tiles -> 4 WMMAs
  v8f o0 = {}, o1 = {};
  #pragma unroll
  for (int kstep = 0; kstep < 2; ++kstep) {
    const _Float16* prow = &pl[wvq][ln][kstep * 32];
    v8h plo = *(const v8h*)(prow + 16 * lh);
    v8h phi = *(const v8h*)(prow + 16 * lh + 8);
    v16h ap;
    #pragma unroll
    for (int j = 0; j < 8; ++j) { ap[j] = plo[j]; ap[j + 8] = phi[j]; }
    v16h bv0 = *(const v16h*)&vlt[ln     ][kstep * 32 + 16 * lh];
    v16h bv1 = *(const v16h*)&vlt[16 + ln][kstep * 32 + 16 * lh];
    o0 = wmma_f16(ap, bv0, o0);
    o1 = wmma_f16(ap, bv1, o1);
  }

  #pragma unroll
  for (int r = 0; r < 8; ++r) {
    const int qq = wvq * 16 + r + 8 * lh;
    if (qq < WIN_) {
      const int qy = qq / WS_, qx = qq % WS_;
      _Float16* dst = aout
          + ((size_t)bh * NPIX + (size_t)(wy * WS_ + qy) * HW_ + (wx * WS_ + qx)) * DH_;
      dst[ln]      = (_Float16)o0[r];
      dst[16 + ln] = (_Float16)o1[r];
    }
  }
}

// ---------------------------------------------------------------------------
// Kernel 4: out = w_proj (96x96) @ attn_out + b_proj (f32 output)
// Block: w_proj staged to LDS f16 once; each wave owns one N-tile and
// iterates all 6 M-tiles with the B fragment held in registers
// (attn_out read exactly once).
// ---------------------------------------------------------------------------
__global__ void __launch_bounds__(256) proj_kernel(
    const _Float16* __restrict__ aout, const float* __restrict__ w_proj,
    const float* __restrict__ b_proj, float* __restrict__ out)
{
  __shared__ _Float16 wlds[96][96];

  const int tid = threadIdx.x;
  {
    _Float16* wd = &wlds[0][0];
    for (int i = tid; i < 96 * 96; i += 256) wd[i] = (_Float16)w_proj[i];
  }
  __syncthreads();

  const int lane = tid & 31;
  const int wv   = tid >> 5;
  const int ln = lane & 15, lh = lane >> 4;
  const int nt = blockIdx.x * 8 + wv;
  const int p  = nt * 16 + ln;
  const int b  = p / NPIX, yx = p % NPIX;

  v8f acc[6] = {{}, {}, {}, {}, {}, {}};
  #pragma unroll
  for (int ks = 0; ks < 3; ++ks) {
    // channel block ks*32 + 16*lh + j -> head ks, dch 16*lh+j : contiguous
    const _Float16* brow = aout + ((size_t)(b * HEADS_ + ks) * NPIX + yx) * DH_ + 16 * lh;
    v16h bm = *(const v16h*)brow;
    #pragma unroll
    for (int mtl = 0; mtl < 6; ++mtl) {
      const _Float16* ar = &wlds[mtl * 16 + ln][ks * 32];
      v8h alo = *(const v8h*)(ar + 16 * lh);
      v8h ahi = *(const v8h*)(ar + 16 * lh + 8);
      v16h a;
      #pragma unroll
      for (int j = 0; j < 8; ++j) { a[j] = alo[j]; a[j + 8] = ahi[j]; }
      acc[mtl] = wmma_f16(a, bm, acc[mtl]);
    }
  }

  #pragma unroll
  for (int mtl = 0; mtl < 6; ++mtl) {
    const int obase = mtl * 16 + lh * 8;
    #pragma unroll
    for (int r = 0; r < 8; ++r)
      out[((size_t)b * DIM_ + (obase + r)) * NPIX + yx] = acc[mtl][r] + b_proj[obase + r];
  }
}

// ---------------------------------------------------------------------------
extern "C" void kernel_launch(void* const* d_in, const int* in_sizes, int n_in,
                              void* d_out, int out_size, void* d_ws, size_t ws_size,
                              hipStream_t stream) {
  const float* x      = (const float*)d_in[0];
  const float* w_qkv  = (const float*)d_in[1];
  const float* b_qkv  = (const float*)d_in[2];
  const float* w_off  = (const float*)d_in[3];
  const float* b_off  = (const float*)d_in[4];
  const float* w_scl  = (const float*)d_in[5];
  const float* b_scl  = (const float*)d_in[6];
  const float* rpb    = (const float*)d_in[7];
  const float* w_proj = (const float*)d_in[8];
  const float* b_proj = (const float*)d_in[9];
  float* out = (float*)d_out;

  const size_t QSZ = (size_t)B_ * HEADS_ * NPIX * DH_;     // 19,267,584 halves
  _Float16* qbuf = (_Float16*)d_ws;
  _Float16* kbuf = qbuf + QSZ;
  _Float16* vbuf = kbuf + QSZ;
  _Float16* aobf = vbuf + QSZ;
  float* offBuf  = (float*)(aobf + QSZ);
  float* sclBuf  = offBuf + (size_t)B_ * HEADS_ * 2 * WN_ * WN_;

  qkv_kernel <<<3 * (NTILES / 8), 256, 0, stream>>>(x, w_qkv, b_qkv, qbuf, kbuf, vbuf);
  pool_off_scl_kernel<<<NWIN_, DIM_, 0, stream>>>(x, w_off, b_off, w_scl, b_scl, offBuf, sclBuf);
  attn_kernel<<<NWIN_ * HEADS_, 128, 0, stream>>>(qbuf, kbuf, vbuf, offBuf, sclBuf, rpb, aobf);
  proj_kernel<<<NTILES / 8, 256, 0, stream>>>(aobf, w_proj, b_proj, out);
}
